// AttentionBlock_83141976916672
// MI455X (gfx1250) — matrix operand order repaired and checked by reading, not compiled, not run on hardware
//
#include <hip/hip_runtime.h>
#include <cstdint>
#include <cstddef>

// ---------------------------------------------------------------------------
// Types for CDNA5 WMMA (wave32, 16x16x32 f16 -> f32)
// ---------------------------------------------------------------------------
typedef _Float16 h8  __attribute__((ext_vector_type(8)));
typedef _Float16 h16 __attribute__((ext_vector_type(16)));
typedef float    f8v __attribute__((ext_vector_type(8)));

#define HW   4096    // H * W pixels per batch image
#define COLS 16384   // N * H * W = 4 * 64 * 64 columns in (C, N*HW) layout
#define LDT  40      // LDS tile stride in halves (multiple of 8 -> 16B rows)

static __device__ __forceinline__ h16 hcat(h8 lo, h8 hi) {
  return __builtin_shufflevector(lo, hi, 0,1,2,3,4,5,6,7,8,9,10,11,12,13,14,15);
}
static __device__ __forceinline__ float sigmoidf_(float x) {
  return 1.f / (1.f + __expf(-x));
}
static __device__ __forceinline__ float eluf_(float x) {
  return (x > 0.f) ? x : expm1f(x);
}

// ---------------------------------------------------------------------------
// Generic tiled WMMA GEMM:  C[M x N] = A[M x K] * B[K x N] (+ bias[m]) (+= C)
//   A: f16 row-major, leading dim lda (K direction), rows >= M guarded
//   B: f16 row-major, leading dim ldb (N direction); rows 0..K-1 must be valid
//   C: f32, leading dim ldc
//   N (gridDim.x*64) and K must be multiples of 64 / 32 respectively.
// Block: 256 threads = 8 waves; each wave computes two 16x16 C tiles.
// ---------------------------------------------------------------------------
__global__ __launch_bounds__(256) void k_gemm_wmma(
    const _Float16* __restrict__ A, int lda,
    const _Float16* __restrict__ B, long ldb,
    float* __restrict__ C, long ldc,
    const float* __restrict__ bias,
    int M, int K, int accumulate)
{
  __shared__ __align__(16) _Float16 As[64 * LDT];   // [m][k]
  __shared__ __align__(16) _Float16 Bs[64 * LDT];   // [n][k]  (transposed stage)

  const int tid  = threadIdx.x;
  const int wave = tid >> 5;
  const int lane = tid & 31;
  const int lm   = lane & 15;
  const int hh   = lane >> 4;
  const int col0 = blockIdx.x * 64;
  const int row0 = blockIdx.y * 64;
  const int mi   = wave >> 1;          // C-tile row index (0..3)
  const int ni0  = (wave & 1) * 2;     // first C-tile col index

  // A stage mapping: thread -> 8 contiguous halves of one A row
  const int am = tid >> 2;             // 0..63
  const int ak = (tid & 3) * 8;        // 0,8,16,24
  // B stage mapping: thread -> 8 contiguous columns of one B k-row
  const int bk  = tid >> 3;            // 0..31
  const int bn0 = (tid & 7) * 8;       // 0..56

  f8v acc[2] = {};

  for (int k0 = 0; k0 < K; k0 += 32) {
    // --- stage A tile (row-guarded; zero-fill beyond M) ---
    h8 av = {};
    const int ga = row0 + am;
    if (ga < M) av = *(const h8*)(A + (size_t)ga * lda + (size_t)(k0 + ak));
    *(h8*)(&As[am * LDT + ak]) = av;

    // --- stage B tile with transpose to [n][k] ---
    h8 bv = *(const h8*)(B + (size_t)(k0 + bk) * ldb + (size_t)(col0 + bn0));
    #pragma unroll
    for (int j = 0; j < 8; ++j) Bs[(bn0 + j) * LDT + bk] = bv[j];

    if (k0 + 32 < K)  // prefetch next B tile (global_prefetch)
      __builtin_prefetch((const void*)(B + (size_t)(k0 + 32 + bk) * ldb +
                                       (size_t)(col0 + bn0)), 0, 1);
    __syncthreads();

    // --- A fragment: lane holds row m=lm, k in {8h..8h+7} U {16+8h..16+8h+7}
    const h8 a0 = *(const h8*)(&As[(mi * 16 + lm) * LDT + 8 * hh]);
    const h8 a1 = *(const h8*)(&As[(mi * 16 + lm) * LDT + 16 + 8 * hh]);
    const h16 afrag = hcat(a0, a1);

    // --- B fragments: lane holds col n=lm, k in {8h..8h+7} U {16+8h..16+8h+7} ---
    {
      const h8 b0 = *(const h8*)(&Bs[((ni0    ) * 16 + lm) * LDT + 8 * hh]);
      const h8 b1 = *(const h8*)(&Bs[((ni0    ) * 16 + lm) * LDT + 16 + 8 * hh]);
      acc[0] = __builtin_amdgcn_wmma_f32_16x16x32_f16(
          false, afrag, false, hcat(b0, b1), (short)0, acc[0], false, false);
    }
    {
      const h8 b0 = *(const h8*)(&Bs[((ni0 + 1) * 16 + lm) * LDT + 8 * hh]);
      const h8 b1 = *(const h8*)(&Bs[((ni0 + 1) * 16 + lm) * LDT + 16 + 8 * hh]);
      acc[1] = __builtin_amdgcn_wmma_f32_16x16x32_f16(
          false, afrag, false, hcat(b0, b1), (short)0, acc[1], false, false);
    }
    __syncthreads();
  }

  // --- epilogue: lane holds C[r + 8h][col] for r=0..7 ---
  #pragma unroll
  for (int q = 0; q < 2; ++q) {
    const int col = col0 + (ni0 + q) * 16 + lm;
    #pragma unroll
    for (int r = 0; r < 8; ++r) {
      const int m = row0 + mi * 16 + 8 * hh + r;
      if (m < M) {
        float val = acc[q][r];
        if (bias) val += bias[m];
        const size_t off = (size_t)m * ldc + col;
        if (accumulate) val += C[off];
        C[off] = val;
      }
    }
  }
}

// ---------------------------------------------------------------------------
// Weight-norm prep: wT16[o][i] = (h)( v[i][o] * g[i] / ||v[i,:]|| ), padded.
// One block per input row i.
// ---------------------------------------------------------------------------
__global__ void k_prep_w(const float* __restrict__ v, const float* __restrict__ g,
                         _Float16* __restrict__ wT, int din, int dout, int ldk)
{
  __shared__ float red[256];
  const int i = blockIdx.x;
  const int t = threadIdx.x;
  float s = 0.f;
  for (int o = t; o < dout; o += 256) {
    float x = v[(size_t)i * dout + o];
    s += x * x;
  }
  red[t] = s; __syncthreads();
  for (int st = 128; st > 0; st >>= 1) { if (t < st) red[t] += red[t + st]; __syncthreads(); }
  const float scale = g[i] * rsqrtf(red[0]);
  for (int o = t; o < dout; o += 256)
    wT[(size_t)o * ldk + i] = (_Float16)(v[(size_t)i * dout + o] * scale);
}

// ---------------------------------------------------------------------------
// concat_elu from up to 3 NCHW fp32 sources -> f16 (rowsPad x COLS), zero pad.
// grid = (64, rowsPad); rows [0,Ctot)=elu(src), [Ctot,2Ctot)=elu(-src).
// ---------------------------------------------------------------------------
__global__ void k_concat_elu_nchw(const float* __restrict__ s0, int c0,
                                  const float* __restrict__ s1, int c1,
                                  const float* __restrict__ s2, int c2,
                                  _Float16* __restrict__ E, int Ctot)
{
  const int col = blockIdx.x * 256 + threadIdx.x;
  const int c   = blockIdx.y;
  float e = 0.f;
  if (c < 2 * Ctot) {
    const int   cc  = (c < Ctot) ? c : c - Ctot;
    const float sgn = (c < Ctot) ? 1.f : -1.f;
    const int n = col >> 12, p = col & 4095;
    float x;
    if (cc < c0)            x = s0[((size_t)n * c0 + cc) * HW + p];
    else if (cc < c0 + c1)  x = s1[((size_t)n * c1 + (cc - c0)) * HW + p];
    else                    x = s2[((size_t)n * c2 + (cc - c0 - c1)) * HW + p];
    e = eluf_(sgn * x);
  }
  E[(size_t)c * COLS + col] = (_Float16)e;
}

// concat_elu from f32 (C x COLS) buffer -> f16 (rowsPad x COLS)
__global__ void k_concat_elu_buf(const float* __restrict__ T,
                                 _Float16* __restrict__ E, int Ctot)
{
  const int col = blockIdx.x * 256 + threadIdx.x;
  const int c   = blockIdx.y;
  float e = 0.f;
  if (c < 2 * Ctot) {
    const int   cc  = (c < Ctot) ? c : c - Ctot;
    const float sgn = (c < Ctot) ? 1.f : -1.f;
    e = eluf_(sgn * T[(size_t)cc * COLS + col]);
  }
  E[(size_t)c * COLS + col] = (_Float16)e;
}

// GRN gate: G16[c][col] = orig(NCHW cat) + aa*sigmoid(bb), zero-padded rows.
__global__ void k_grn_gate_f16(const float* __restrict__ T2,
                               const float* __restrict__ s0, int c0,
                               const float* __restrict__ s1, int c1,
                               const float* __restrict__ s2, int c2,
                               _Float16* __restrict__ G, int Ctot)
{
  const int col = blockIdx.x * 256 + threadIdx.x;
  const int c   = blockIdx.y;
  float g = 0.f;
  if (c < Ctot) {
    const int n = col >> 12, p = col & 4095;
    float orig;
    if (c < c0)            orig = s0[((size_t)n * c0 + c) * HW + p];
    else if (c < c0 + c1)  orig = s1[((size_t)n * c1 + (c - c0)) * HW + p];
    else                   orig = s2[((size_t)n * c2 + (c - c0 - c1)) * HW + p];
    const float aa = T2[(size_t)c * COLS + col];
    const float bb = T2[(size_t)(c + Ctot) * COLS + col];
    g = orig + aa * sigmoidf_(bb);
  }
  G[(size_t)c * COLS + col] = (_Float16)g;
}

// Final gate: out(NCHW) = ul + aa*sigmoid(bb). grid = (64, 160)
__global__ void k_grn_gate_out(const float* __restrict__ T2,
                               const float* __restrict__ ul,
                               float* __restrict__ out)
{
  const int col = blockIdx.x * 256 + threadIdx.x;
  const int c   = blockIdx.y;
  const int n = col >> 12, p = col & 4095;
  const float orig = ul[((size_t)n * 160 + c) * HW + p];
  const float aa = T2[(size_t)c * COLS + col];
  const float bb = T2[(size_t)(c + 160) * COLS + col];
  out[((size_t)n * 160 + c) * HW + p] = orig + aa * sigmoidf_(bb);
}

// f32 (rows x COLS) -> f16 (rowsPad x COLS), zero pad. grid=(64, rowsPad)
__global__ void k_cast_rows_f16(const float* __restrict__ S,
                                _Float16* __restrict__ D, int rows)
{
  const int col = blockIdx.x * 256 + threadIdx.x;
  const int r   = blockIdx.y;
  const float x = (r < rows) ? S[(size_t)r * COLS + col] : 0.f;
  D[(size_t)r * COLS + col] = (_Float16)x;
}

// queries (16 x COLS) f32 -> qT (COLS x 32) f16 with K-pad zeros
__global__ void k_qT(const float* __restrict__ Q, _Float16* __restrict__ QT)
{
  const int col = blockIdx.x * 256 + threadIdx.x;
  #pragma unroll
  for (int d = 0; d < 16; ++d)
    QT[(size_t)col * 32 + d] = (_Float16)Q[(size_t)d * COLS + col];
  #pragma unroll
  for (int d = 16; d < 32; ++d)
    QT[(size_t)col * 32 + d] = (_Float16)0.f;
}

// Multiplicative-mask softmax, exactly matching reference semantics:
// val = (i > q) ? S[q,i] : 0 ; softmax over i (zeros participate: exp(0)=1).
// One block (256 threads) per row q; 4096 cols -> 16 per thread.
__global__ void k_softmax_mask(const float* __restrict__ S,
                               _Float16* __restrict__ W)
{
  __shared__ float red[256];
  const int q = blockIdx.x;
  const int t = threadIdx.x;
  float v[16];
  float mx = 0.f;   // a masked 0 is always present in every row
  #pragma unroll
  for (int j = 0; j < 16; ++j) {
    const int i = t + 256 * j;
    const float x = (i > q) ? S[(size_t)q * 4096 + i] : 0.f;
    v[j] = x; mx = fmaxf(mx, x);
  }
  red[t] = mx; __syncthreads();
  for (int s = 128; s > 0; s >>= 1) { if (t < s) red[t] = fmaxf(red[t], red[t + s]); __syncthreads(); }
  mx = red[0]; __syncthreads();
  float sum = 0.f;
  #pragma unroll
  for (int j = 0; j < 16; ++j) { v[j] = __expf(v[j] - mx); sum += v[j]; }
  red[t] = sum; __syncthreads();
  for (int s = 128; s > 0; s >>= 1) { if (t < s) red[t] += red[t + s]; __syncthreads(); }
  const float inv = 1.f / red[0];
  #pragma unroll
  for (int j = 0; j < 16; ++j) {
    const int i = t + 256 * j;
    W[(size_t)q * 4096 + i] = (_Float16)(v[j] * inv);
  }
}

// ---------------------------------------------------------------------------
// Host side
// ---------------------------------------------------------------------------
struct NinPar { const float* v; const float* g; const float* b; };

extern "C" void kernel_launch(void* const* d_in, const int* in_sizes, int n_in,
                              void* d_out, int out_size, void* d_ws, size_t ws_size,
                              hipStream_t stream)
{
  (void)out_size; (void)ws_size; (void)n_in;

  // ---- map inputs: insertion order (x first) vs tree_flatten alpha (b first)
  const float *x, *ul, *bimg;
  NinPar P[12];   // 0 gk.ci 1 gk.co 2 gq.ci 3 gq.co 4 gv.ci 5 gv.co
                  // 6 nin_k 7 nin_q 8 nin_v 9 go.ci 10 go.co 11 go.skip
  const bool alpha = (in_sizes[0] == 98304);   // 'b' image first when sorted
  if (!alpha) {
    x = (const float*)d_in[0]; ul = (const float*)d_in[1]; bimg = (const float*)d_in[2];
    for (int L = 0; L < 12; ++L) {
      P[L].v = (const float*)d_in[3 + 3 * L];
      P[L].g = (const float*)d_in[4 + 3 * L];
      P[L].b = (const float*)d_in[5 + 3 * L];
    }
  } else {
    bimg = (const float*)d_in[0]; ul = (const float*)d_in[37]; x = (const float*)d_in[38];
    // sorted nin order: gk.ci gk.co go.ci go.co go.skip gq.ci gq.co gv.ci gv.co nk nq nv
    const int mapP[12] = {0, 1, 5, 6, 7, 8, 9, 10, 11, 2, 3, 4};
    for (int L = 0; L < 12; ++L) {
      const int base = 1 + 3 * mapP[L];        // leaves sorted: b, g, v
      P[L].b = (const float*)d_in[base];
      P[L].g = (const float*)d_in[base + 1];
      P[L].v = (const float*)d_in[base + 2];
    }
  }

  // ---- layer dims
  const int din_[12]  = {338,338,332,332,338,338,169,166,169,320,320,160};
  const int dout_[12] = {169,338,166,332,169,338, 16, 16, 80,160,320,160};
  const int ldk_[12]  = {352,352,352,352,352,352,192,192,192,320,320,160};
  const int mpad_[12] = {176,352,176,336,176,352, 16, 16, 80,160,320,160};

  // ---- workspace carve-up (256B aligned slices)
  size_t off = 0;
  char* wsb = (char*)d_ws;
  auto take = [&](size_t bytes) -> char* {
    char* p = wsb + off;
    off += (bytes + 255) & ~(size_t)255;
    return p;
  };
  _Float16* WT[12];
  size_t wtBytes[12];
  for (int L = 0; L < 12; ++L) {
    wtBytes[L] = (size_t)mpad_[L] * ldk_[L] * sizeof(_Float16);
    WT[L] = (_Float16*)take(wtBytes[L]);
  }
  _Float16* EBUF = (_Float16*)take((size_t)352 * COLS * 2);
  float*    TBUF = (float*)   take((size_t)352 * COLS * 4);
  _Float16* G16  = (_Float16*)take((size_t)192 * COLS * 2);
  float*    KEYS = (float*)   take((size_t)16  * COLS * 4);
  float*    QUER = (float*)   take((size_t)16  * COLS * 4);
  float*    VALS = (float*)   take((size_t)80  * COLS * 4);
  _Float16* K16  = (_Float16*)take((size_t)32  * COLS * 2);
  _Float16* QT16 = (_Float16*)take((size_t)COLS * 32 * 2);
  _Float16* V16  = (_Float16*)take((size_t)80  * COLS * 2);
  float*    SC   = (float*)   take((size_t)4096 * 4096 * 4);
  _Float16* WSM  = (_Float16*)take((size_t)4096 * 4096 * 2);
  float*    WVAL = (float*)   take((size_t)80  * COLS * 4);

  auto gemm = [&](const _Float16* A, int lda, const _Float16* B, long ldb,
                  float* C, long ldc, const float* bias,
                  int M, int N, int K, int acc) {
    dim3 g(N / 64, (M + 63) / 64);
    k_gemm_wmma<<<g, 256, 0, stream>>>(A, lda, B, ldb, C, ldc, bias, M, K, acc);
  };

  // ---- weight normalization (zero-pad arena, then fill)
  for (int L = 0; L < 12; ++L) (void)hipMemsetAsync(WT[L], 0, wtBytes[L], stream);
  for (int L = 0; L < 12; ++L)
    k_prep_w<<<din_[L], 256, 0, stream>>>(P[L].v, P[L].g, WT[L], din_[L], dout_[L], ldk_[L]);

  // ---- grn_k(x_ul_b) -> nin_k -> keys (16 x COLS)
  k_concat_elu_nchw<<<dim3(64,352),256,0,stream>>>(x,3, ul,160, bimg,6, EBUF, 169);
  gemm(WT[0],352, EBUF,COLS, TBUF,COLS, P[0].b, 169, COLS, 352, 0);
  k_concat_elu_buf<<<dim3(64,352),256,0,stream>>>(TBUF, EBUF, 169);
  gemm(WT[1],352, EBUF,COLS, TBUF,COLS, P[1].b, 338, COLS, 352, 0);
  k_grn_gate_f16<<<dim3(64,192),256,0,stream>>>(TBUF, x,3, ul,160, bimg,6, G16, 169);
  gemm(WT[6],192, G16,COLS, KEYS,COLS, P[6].b, 16, COLS, 192, 0);

  // ---- grn_q(ul_b) -> nin_q -> queries (16 x COLS)
  k_concat_elu_nchw<<<dim3(64,352),256,0,stream>>>(ul,160, bimg,6, nullptr,0, EBUF, 166);
  gemm(WT[2],352, EBUF,COLS, TBUF,COLS, P[2].b, 166, COLS, 352, 0);
  k_concat_elu_buf<<<dim3(64,352),256,0,stream>>>(TBUF, EBUF, 166);
  gemm(WT[3],352, EBUF,COLS, TBUF,COLS, P[3].b, 332, COLS, 352, 0);
  k_grn_gate_f16<<<dim3(64,192),256,0,stream>>>(TBUF, ul,160, bimg,6, nullptr,0, G16, 166);
  gemm(WT[7],192, G16,COLS, QUER,COLS, P[7].b, 16, COLS, 192, 0);

  // ---- grn_v(x_ul_b) -> nin_v -> values (80 x COLS)
  k_concat_elu_nchw<<<dim3(64,352),256,0,stream>>>(x,3, ul,160, bimg,6, EBUF, 169);
  gemm(WT[4],352, EBUF,COLS, TBUF,COLS, P[4].b, 169, COLS, 352, 0);
  k_concat_elu_buf<<<dim3(64,352),256,0,stream>>>(TBUF, EBUF, 169);
  gemm(WT[5],352, EBUF,COLS, TBUF,COLS, P[5].b, 338, COLS, 352, 0);
  k_grn_gate_f16<<<dim3(64,192),256,0,stream>>>(TBUF, x,3, ul,160, bimg,6, G16, 169);
  gemm(WT[8],192, G16,COLS, VALS,COLS, P[8].b, 80, COLS, 192, 0);

  // ---- attention: per-batch scores (K=16 pad 32), mask-softmax, A*V
  k_cast_rows_f16<<<dim3(64,32),256,0,stream>>>(KEYS, K16, 16);
  k_qT<<<64,256,0,stream>>>(QUER, QT16);
  k_cast_rows_f16<<<dim3(64,80),256,0,stream>>>(VALS, V16, 80);
  for (int n = 0; n < 4; ++n) {
    // S[q,i] = sum_d Q[d, n*HW+q] * K[d, n*HW+i]
    gemm(QT16 + (size_t)n * 4096 * 32, 32, K16 + (size_t)n * 4096, COLS,
         SC, 4096, nullptr, 4096, 4096, 32, 0);
    k_softmax_mask<<<4096,256,0,stream>>>(SC, WSM);
    // WVAL[v, n*HW+i] = sum_q V[v, n*HW+q] * W[q,i]
    gemm(V16 + (size_t)n * 4096, COLS, WSM, 4096,
         WVAL + (size_t)n * 4096, COLS, nullptr, 80, 4096, 4096, 0);
  }

  // ---- grn_out(ul, skip=w_values) -> d_out (NCHW f32)
  k_concat_elu_nchw<<<dim3(64,320),256,0,stream>>>(ul,160, nullptr,0, nullptr,0, EBUF, 160);
  gemm(WT[9],320, EBUF,COLS, TBUF,COLS, P[9].b, 160, COLS, 320, 0);
  k_concat_elu_buf<<<dim3(64,160),256,0,stream>>>(WVAL, EBUF, 80);
  gemm(WT[11],160, EBUF,COLS, TBUF,COLS, P[11].b, 160, COLS, 160, 1);  // += skip
  k_concat_elu_buf<<<dim3(64,320),256,0,stream>>>(TBUF, EBUF, 160);
  gemm(WT[10],320, EBUF,COLS, TBUF,COLS, P[10].b, 320, COLS, 320, 0);
  k_grn_gate_out<<<dim3(64,160),256,0,stream>>>(TBUF, ul, (float*)d_out);
}
